// PrenormEncoderLayer_63943473103395
// MI455X (gfx1250) — compile-verified, matrix operand order checked by reading, not run on hardware
//
#include <hip/hip_runtime.h>
#include <hip/hip_bf16.h>
#include <cstdint>

typedef __attribute__((ext_vector_type(16))) _Float16 v16h;
typedef __attribute__((ext_vector_type(8)))  _Float16 v8h;
typedef __attribute__((ext_vector_type(8)))  float    v8f;
typedef __attribute__((ext_vector_type(4)))  unsigned int u32x4;
typedef __attribute__((ext_vector_type(4)))  int      i32x4;
typedef __attribute__((ext_vector_type(8)))  int      i32x8;

#define B_    2
#define N_    4096
#define HID_  768
#define NH_   12
#define DH_   64
#define BLK_  64
#define M_    64
#define R_    3
#define FF_   3072
#define ROWS_ (B_*N_)   // 8192

#ifndef __has_builtin
#define __has_builtin(x) 0
#endif
#if __has_builtin(__builtin_amdgcn_tensor_load_to_lds) && __has_builtin(__builtin_amdgcn_s_wait_tensorcnt)
#define HAVE_TDM 1
#else
#define HAVE_TDM 0
#endif

// ---------------------------------------------------------------------------
// WMMA helpers
// ---------------------------------------------------------------------------
__device__ __forceinline__ v8f wmma_f16(v16h a, v16h b, v8f c) {
  return __builtin_amdgcn_wmma_f32_16x16x32_f16(
      false, a, false, b, (short)0, c, false, false);
}

// Fragment loader valid for BOTH A (16x32) and B (32x16 from a transposed
// array): lane l holds row (row0 + (l&15)) elements
// {k0+ofs..+7, k0+ofs+16..+23}, ofs = (l<16)?0:8. Two 16B contiguous loads.
// Works for global (global_load_b128) and LDS (ds_load_b128) pointers.
__device__ __forceinline__ v16h load_frag(const _Float16* __restrict__ p,
                                          int ld, int row0, int k0) {
  const int lane = threadIdx.x & 31;
  const _Float16* q = p + (size_t)(row0 + (lane & 15)) * ld + k0
                        + ((lane < 16) ? 0 : 8);
  v8h lo = *(const v8h*)(q);
  v8h hi = *(const v8h*)(q + 16);
  v16h r;
#pragma unroll
  for (int i = 0; i < 8; ++i) { r[i] = lo[i]; r[i + 8] = hi[i]; }
  return r;
}

__device__ __forceinline__ float gelu_f(float x) {
  return 0.5f * x * (1.0f + tanhf(0.7978845608028654f * (x + 0.044715f * x * x * x)));
}

// ---------------------------------------------------------------------------
// Tensor Data Mover: 2-D tile load (global -> LDS), D# per ISA 08 §8.3-8.4.
// data_size = 2 bytes. Issued by one wave; tracked with TENSORcnt.
// ---------------------------------------------------------------------------
__device__ __forceinline__ unsigned lds_off(const void* p) {
  // ISA §10.2: LDS aperture maps addr[31:0] -> LDS byte offset.
  return (unsigned)(unsigned long long)(uintptr_t)p;
}

#if HAVE_TDM
__device__ __forceinline__ void tdm_load_2d(const void* gptr, unsigned ldsoff,
                                            unsigned tile_w, unsigned tile_h,
                                            unsigned tensor_w, unsigned tensor_h,
                                            unsigned row_stride) {
  unsigned long long ga = (unsigned long long)(uintptr_t)gptr;
  u32x4 g0;
  g0[0] = 1u;                                   // count=1, user mode, no gather
  g0[1] = ldsoff;                               // lds_addr [63:32]
  g0[2] = (unsigned)(ga & 0xFFFFFFFFu);         // global_addr [95:64]
  g0[3] = (unsigned)((ga >> 32) & 0x1FFFFFFu)   // global_addr [120:96]
        | (2u << 30);                           // type=2 ("image") [127:126]
  i32x8 g1;
  g1[0] = (int)(1u << 16);                      // data_size=1 (2 bytes)
  g1[1] = (int)((tensor_w & 0xFFFFu) << 16);    // tensor_dim0[15:0] @ [63:48]
  g1[2] = (int)((tensor_w >> 16) | ((tensor_h & 0xFFFFu) << 16)); // td0 hi | td1 lo
  g1[3] = (int)((tensor_h >> 16) | (tile_w << 16));               // td1 hi | tile_dim0
  g1[4] = (int)(tile_h & 0xFFFFu);              // tile_dim1 (tile_dim2 = 0)
  g1[5] = (int)row_stride;                      // tensor_dim0_stride lo32
  g1[6] = 0;                                    // stride0 hi16 | stride1 lo16
  g1[7] = 0;
  i32x4 z4 = {0, 0, 0, 0};
#if __clang_major__ >= 23
  i32x8 z8 = {0, 0, 0, 0, 0, 0, 0, 0};
  __builtin_amdgcn_tensor_load_to_lds(g0, g1, z4, z4, z8, 0);
#else
  __builtin_amdgcn_tensor_load_to_lds(g0, g1, z4, z4, 0);
#endif
}
#endif

// Stage one 64x64 K tile (row-contiguous) and one 64x64 V tile (64 dims x 64
// keys sliced out of vT rows of length N_) into LDS.
__device__ __forceinline__ void stage_kv(const _Float16* __restrict__ kbh,
                                         const _Float16* __restrict__ vtb,
                                         int krow0,
                                         _Float16* __restrict__ skb,
                                         _Float16* __restrict__ svb,
                                         int wave, int tid) {
#if HAVE_TDM
  if (wave == 0) {
    tdm_load_2d(kbh + (size_t)krow0 * DH_, lds_off(skb), DH_, BLK_, DH_, N_, DH_);
    tdm_load_2d(vtb + krow0,               lds_off(svb), BLK_, DH_, N_, DH_, N_);
  }
#else
#pragma unroll
  for (int it = 0; it < 4; ++it) {
    int c = tid + it * 128;                       // 512 x 16B chunks
    *(v8h*)(skb + c * 8) = *(const v8h*)(kbh + (size_t)krow0 * DH_ + c * 8);
    int r = c >> 3, cc = (c & 7) * 8;
    *(v8h*)(svb + r * BLK_ + cc) = *(const v8h*)(vtb + (size_t)r * N_ + krow0 + cc);
  }
#endif
}

// ---------------------------------------------------------------------------
// Weight / bias packing (f32 -> f16, transposed so B operand rows are contiguous)
// ---------------------------------------------------------------------------
__global__ __launch_bounds__(256) void pack_wqkv(const float* __restrict__ wq,
                                                 const float* __restrict__ wk,
                                                 const float* __restrict__ wv,
                                                 _Float16* __restrict__ dst) {
  int idx = blockIdx.x * 256 + threadIdx.x;           // [2304][768]
  if (idx >= 3 * HID_ * HID_) return;
  int d = idx % HID_;
  int c = idx / HID_;
  int seg = c / HID_;
  int cc = c % HID_;
  const float* s = (seg == 0) ? wq : (seg == 1) ? wk : wv;
  dst[idx] = (_Float16)s[(size_t)d * HID_ + cc];
}

__global__ __launch_bounds__(256) void transpose_w(const float* __restrict__ src,
                                                   _Float16* __restrict__ dst,
                                                   int K, int Nc) {
  int idx = blockIdx.x * 256 + threadIdx.x;
  if (idx >= K * Nc) return;
  int k = idx % K;
  int n = idx / K;
  dst[idx] = (_Float16)src[(size_t)k * Nc + n];
}

__global__ __launch_bounds__(256) void pack_qkv_bias(const float* __restrict__ qb,
                                                     const float* __restrict__ kb,
                                                     const float* __restrict__ vb,
                                                     float* __restrict__ dst) {
  int i = blockIdx.x * 256 + threadIdx.x;
  if (i >= 3 * HID_) return;
  const float* s = (i < HID_) ? qb : (i < 2 * HID_) ? kb : vb;
  dst[i] = s[i % HID_];
}

// ---------------------------------------------------------------------------
// LayerNorm (f32 in -> f16 out), one block per row of 768
// ---------------------------------------------------------------------------
__global__ __launch_bounds__(256) void layernorm_f16_kernel(
    const float* __restrict__ x, const float* __restrict__ g,
    const float* __restrict__ be, _Float16* __restrict__ out) {
  const int row = blockIdx.x, tid = threadIdx.x;
  const float* xr = x + (size_t)row * HID_;
  float v0 = xr[tid], v1 = xr[tid + 256], v2 = xr[tid + 512];
  __shared__ float red[256];
  red[tid] = v0 + v1 + v2;
  __syncthreads();
  for (int s = 128; s > 0; s >>= 1) { if (tid < s) red[tid] += red[tid + s]; __syncthreads(); }
  float mu = red[0] * (1.0f / HID_);
  __syncthreads();
  float d0 = v0 - mu, d1 = v1 - mu, d2 = v2 - mu;
  red[tid] = d0 * d0 + d1 * d1 + d2 * d2;
  __syncthreads();
  for (int s = 128; s > 0; s >>= 1) { if (tid < s) red[tid] += red[tid + s]; __syncthreads(); }
  float rstd = rsqrtf(red[0] * (1.0f / HID_) + 1e-12f);
  _Float16* o = out + (size_t)row * HID_;
  o[tid]       = (_Float16)(d0 * rstd * g[tid]       + be[tid]);
  o[tid + 256] = (_Float16)(d1 * rstd * g[tid + 256] + be[tid + 256]);
  o[tid + 512] = (_Float16)(d2 * rstd * g[tid + 512] + be[tid + 512]);
}

// ---------------------------------------------------------------------------
// Repack QKV GEMM output [8192][2304] -> q/k [b][h][n][64] f16, vT [b][h][64][n]
// ---------------------------------------------------------------------------
__global__ __launch_bounds__(256) void repack_qkv(const _Float16* __restrict__ t,
                                                  _Float16* __restrict__ q,
                                                  _Float16* __restrict__ k,
                                                  _Float16* __restrict__ vT) {
  int idx = blockIdx.x * 256 + threadIdx.x;
  if (idx >= B_ * NH_ * N_ * DH_) return;
  int d  = idx & 63;
  int n  = (idx >> 6) & (N_ - 1);
  int bh = idx >> 18;
  int h  = bh % NH_;
  int b  = bh / NH_;
  size_t src = (size_t)(b * N_ + n) * (3 * HID_) + h * DH_ + d;
  q[idx] = t[src];
  k[idx] = t[src + HID_];
  vT[((size_t)(b * NH_ + h) * DH_ + d) * N_ + n] = t[src + 2 * HID_];
}

// ---------------------------------------------------------------------------
// Generic WMMA GEMM: C[M x N] = A_f16[M x K] * Bt_f16[N x K]^T  (+ epilogue)
// 128 threads = 4 waves; block tile 128x128, wave tile 64x64.
// EPI: 0 = +bias -> f16 ; 1 = +bias +resid(f32) -> f32 ; 2 = gelu(+bias) -> f16
// ---------------------------------------------------------------------------
template <int EPI>
__global__ __launch_bounds__(128) void gemm_wmma(
    const _Float16* __restrict__ A, int lda,
    const _Float16* __restrict__ Bt, int ldb,
    const float* __restrict__ bias,
    const float* __restrict__ resid,
    _Float16* __restrict__ outh, float* __restrict__ outf,
    int Ncols, int K) {
  const int lane = threadIdx.x & 31;
  const int wave = threadIdx.x >> 5;
  const int r0 = blockIdx.y * 128 + (wave >> 1) * 64;
  const int c0 = blockIdx.x * 128 + (wave & 1) * 64;

  v8f acc[4][4] = {};
  for (int k0 = 0; k0 < K; k0 += 32) {
    v16h a[4], bfr[4];
#pragma unroll
    for (int mi = 0; mi < 4; ++mi) a[mi]   = load_frag(A,  lda, r0 + 16 * mi, k0);
#pragma unroll
    for (int ni = 0; ni < 4; ++ni) bfr[ni] = load_frag(Bt, ldb, c0 + 16 * ni, k0);
#pragma unroll
    for (int mi = 0; mi < 4; ++mi)
#pragma unroll
      for (int ni = 0; ni < 4; ++ni)
        acc[mi][ni] = wmma_f16(a[mi], bfr[ni], acc[mi][ni]);
  }

  const int colb = lane & 15;
  const int rowb = (lane >> 4) << 3;
#pragma unroll
  for (int mi = 0; mi < 4; ++mi) {
#pragma unroll
    for (int ni = 0; ni < 4; ++ni) {
      const int col = c0 + 16 * ni + colb;
      const int row = r0 + 16 * mi + rowb;
      const float bv = bias[col];
#pragma unroll
      for (int j = 0; j < 8; ++j) {
        float v = acc[mi][ni][j] + bv;
        size_t o = (size_t)(row + j) * Ncols + col;
        if (EPI == 0)      outh[o] = (_Float16)v;
        else if (EPI == 2) outh[o] = (_Float16)gelu_f(v);
        else               outf[o] = v + resid[o];
      }
    }
  }
}

// ---------------------------------------------------------------------------
// BigBird attention: flash-style streaming softmax with TDM double-buffered
// K/V tile staging in LDS. grid = (M_, NH_, B_), 128 threads = 4 waves.
// ---------------------------------------------------------------------------
__global__ __launch_bounds__(128) void bigbird_attn_kernel(
    const _Float16* __restrict__ q, const _Float16* __restrict__ k,
    const _Float16* __restrict__ vT, const int* __restrict__ rand_attn,
    const float* __restrict__ to_mask, const float* __restrict__ from_mask,
    _Float16* __restrict__ ctx) {
  const int i = blockIdx.x, h = blockIdx.y, b = blockIdx.z;

  __shared__ int s_kb[64];
  __shared__ int s_nkb;
  __shared__ __align__(16) _Float16 s_k[2][BLK_ * DH_];   // 2 x 8 KB
  __shared__ __align__(16) _Float16 s_v[2][DH_ * BLK_];   // 2 x 8 KB
  __shared__ __align__(16) _Float16 s_p[4][16 * 64];      // 8 KB

  if (threadIdx.x == 0) {
    int n = 0;
    if (i == 0 || i == M_ - 1) {
      for (int t = 0; t < M_; ++t) s_kb[n++] = t;          // global rows
    } else {
      s_kb[n++] = 0;
      if (i - 1 > 0)      s_kb[n++] = i - 1;               // sliding window
      s_kb[n++] = i;
      if (i + 1 < M_ - 1) s_kb[n++] = i + 1;
      s_kb[n++] = M_ - 1;
      const int* ra = rand_attn + ((size_t)h * (M_ - 2) + (i - 1)) * R_;
      for (int r = 0; r < R_; ++r) s_kb[n++] = ra[r];      // random blocks
    }
    s_nkb = n;
  }
  __syncthreads();
  const int nkb = s_nkb;

  const int wave = threadIdx.x >> 5, lane = threadIdx.x & 31;
  const _Float16* qbh = q  + (size_t)(b * NH_ + h) * N_ * DH_;
  const _Float16* kbh = k  + (size_t)(b * NH_ + h) * N_ * DH_;
  const _Float16* vtb = vT + (size_t)(b * NH_ + h) * DH_ * N_;
  const int qrow0 = i * BLK_ + wave * 16;

  const v16h qa0 = load_frag(qbh, DH_, qrow0, 0);
  const v16h qa1 = load_frag(qbh, DH_, qrow0, 32);

  v8f o[4] = {};
  float m[8], l[8];
#pragma unroll
  for (int j = 0; j < 8; ++j) { m[j] = -1e30f; l[j] = 0.0f; }
  const float scale = 0.125f;                       // 1/sqrt(64)
  _Float16* sp = &s_p[wave][0];
  const int prow = (lane >> 4) << 3, pcol = lane & 15;

  // prologue: stage block 0 into buffer 0
  stage_kv(kbh, vtb, s_kb[0] * BLK_, &s_k[0][0], &s_v[0][0], wave, threadIdx.x);

  for (int kb = 0; kb < nkb; ++kb) {
    const int cur = kb & 1;
    // stage next block into the other buffer (TDM overlaps with compute)
    if (kb + 1 < nkb)
      stage_kv(kbh, vtb, s_kb[kb + 1] * BLK_, &s_k[cur ^ 1][0], &s_v[cur ^ 1][0],
               wave, threadIdx.x);
#if HAVE_TDM
    if (wave == 0) {
      if (kb + 1 < nkb) __builtin_amdgcn_s_wait_tensorcnt((short)2); // cur done
      else              __builtin_amdgcn_s_wait_tensorcnt((short)0); // drain
    }
#endif
    __syncthreads();                                  // buffer `cur` ready

    const _Float16* skb = &s_k[cur][0];
    const _Float16* svb = &s_v[cur][0];
    const int krow0 = s_kb[kb] * BLK_;

    // --- S = scale * Q K^T + mask (K from LDS) -----------------------------
    v8f s4[4];
#pragma unroll
    for (int t = 0; t < 4; ++t) {
      v16h b0 = load_frag(skb, DH_, t * 16, 0);
      v16h b1 = load_frag(skb, DH_, t * 16, 32);
      v8f z = {};
      z = wmma_f16(qa0, b0, z);
      z = wmma_f16(qa1, b1, z);
      float tm = to_mask[(size_t)b * N_ + krow0 + t * 16 + pcol];
      float madd = (1.0f - tm) * (-10000.0f);
#pragma unroll
      for (int j = 0; j < 8; ++j) z[j] = z[j] * scale + madd;
      s4[t] = z;
    }

    // --- online softmax ----------------------------------------------------
    float nm[8];
#pragma unroll
    for (int j = 0; j < 8; ++j) {
      float v = fmaxf(fmaxf(s4[0][j], s4[1][j]), fmaxf(s4[2][j], s4[3][j]));
      for (int dd = 1; dd < 16; dd <<= 1) v = fmaxf(v, __shfl_xor(v, dd, 32));
      nm[j] = v;
    }
    float rs[8];
#pragma unroll
    for (int j = 0; j < 8; ++j) {
      float mnew = fmaxf(m[j], nm[j]);
      float corr = __expf(m[j] - mnew);
      m[j] = mnew;
      l[j] *= corr;
      o[0][j] *= corr; o[1][j] *= corr; o[2][j] *= corr; o[3][j] *= corr;
      rs[j] = 0.0f;
    }
#pragma unroll
    for (int t = 0; t < 4; ++t) {
#pragma unroll
      for (int j = 0; j < 8; ++j) {
        float p = __expf(s4[t][j] - m[j]);
        rs[j] += p;
        sp[(prow + j) * 64 + t * 16 + pcol] = (_Float16)p;   // P -> A layout
      }
    }
#pragma unroll
    for (int j = 0; j < 8; ++j) {
      float v = rs[j];
      for (int dd = 1; dd < 16; dd <<= 1) v += __shfl_xor(v, dd, 32);
      l[j] += v;
    }
    __syncthreads();                                  // s_p ready

    // --- O += P * V (both operands from LDS) -------------------------------
    v16h pa0 = load_frag(sp, 64, 0, 0);
    v16h pa1 = load_frag(sp, 64, 0, 32);
#pragma unroll
    for (int ot = 0; ot < 4; ++ot) {
      v16h vb0 = load_frag(svb, BLK_, ot * 16, 0);
      v16h vb1 = load_frag(svb, BLK_, ot * 16, 32);
      o[ot] = wmma_f16(pa0, vb0, o[ot]);
      o[ot] = wmma_f16(pa1, vb1, o[ot]);
    }
    // next iteration stages into this buffer before its first barrier; WAR unguarded
  }

  // --- normalize + write ctx [b][n][h*64+e] f16 ----------------------------
#pragma unroll
  for (int j = 0; j < 8; ++j) {
    int n = qrow0 + prow + j;
    float fm = from_mask[(size_t)b * N_ + n];
    float inv = fm / l[j];
    size_t base = (size_t)(b * N_ + n) * HID_ + h * DH_ + pcol;
    ctx[base]      = (_Float16)(o[0][j] * inv);
    ctx[base + 16] = (_Float16)(o[1][j] * inv);
    ctx[base + 32] = (_Float16)(o[2][j] * inv);
    ctx[base + 48] = (_Float16)(o[3][j] * inv);
  }
}

// ---------------------------------------------------------------------------
// Launch
// ---------------------------------------------------------------------------
extern "C" void kernel_launch(void* const* d_in, const int* in_sizes, int n_in,
                              void* d_out, int out_size, void* d_ws, size_t ws_size,
                              hipStream_t stream) {
  (void)in_sizes; (void)n_in; (void)out_size; (void)ws_size;
  const float* x     = (const float*)d_in[0];
  const float* ln1g  = (const float*)d_in[1];
  const float* ln1b  = (const float*)d_in[2];
  const float* wq    = (const float*)d_in[3];
  const float* qbias = (const float*)d_in[4];
  const float* wk    = (const float*)d_in[5];
  const float* kbias = (const float*)d_in[6];
  const float* wv    = (const float*)d_in[7];
  const float* vbias = (const float*)d_in[8];
  const float* wo    = (const float*)d_in[9];
  const float* obias = (const float*)d_in[10];
  const float* ln2g  = (const float*)d_in[11];
  const float* ln2b  = (const float*)d_in[12];
  const float* wff1  = (const float*)d_in[13];
  const float* bff1  = (const float*)d_in[14];
  const float* wff2  = (const float*)d_in[15];
  const float* bff2  = (const float*)d_in[16];
  const int*   ra    = (const int*)d_in[17];
  const float* fromm = (const float*)d_in[19];
  const float* tom   = (const float*)d_in[20];
  float* out = (float*)d_out;

  char* w = (char*)d_ws;
  size_t off = 0;
  auto alloc = [&](size_t bytes) -> void* {
    void* p = w + off;
    off = (off + bytes + 255) & ~(size_t)255;
    return p;
  };
  _Float16* wqkv_t = (_Float16*)alloc((size_t)3 * HID_ * HID_ * 2);
  _Float16* wo_t   = (_Float16*)alloc((size_t)HID_ * HID_ * 2);
  _Float16* wff1_t = (_Float16*)alloc((size_t)FF_ * HID_ * 2);
  _Float16* wff2_t = (_Float16*)alloc((size_t)HID_ * FF_ * 2);
  float*    qkvb   = (float*)alloc((size_t)3 * HID_ * 4);
  _Float16* nx     = (_Float16*)alloc((size_t)ROWS_ * HID_ * 2);   // nx then ny
  _Float16* qf     = (_Float16*)alloc((size_t)B_ * NH_ * N_ * DH_ * 2);
  _Float16* kf     = (_Float16*)alloc((size_t)B_ * NH_ * N_ * DH_ * 2);
  _Float16* vtf    = (_Float16*)alloc((size_t)B_ * NH_ * DH_ * N_ * 2);
  _Float16* ctx    = (_Float16*)alloc((size_t)ROWS_ * HID_ * 2);
  float*    attn   = (float*)alloc((size_t)ROWS_ * HID_ * 4);
  _Float16* bufA   = (_Float16*)alloc((size_t)ROWS_ * FF_ * 2);    // qkv_tmp then h1

  pack_wqkv<<<(3 * HID_ * HID_ + 255) / 256, 256, 0, stream>>>(wq, wk, wv, wqkv_t);
  transpose_w<<<(HID_ * HID_ + 255) / 256, 256, 0, stream>>>(wo, wo_t, HID_, HID_);
  transpose_w<<<(HID_ * FF_ + 255) / 256, 256, 0, stream>>>(wff1, wff1_t, HID_, FF_);
  transpose_w<<<(FF_ * HID_ + 255) / 256, 256, 0, stream>>>(wff2, wff2_t, FF_, HID_);
  pack_qkv_bias<<<(3 * HID_ + 255) / 256, 256, 0, stream>>>(qbias, kbias, vbias, qkvb);

  layernorm_f16_kernel<<<ROWS_, 256, 0, stream>>>(x, ln1g, ln1b, nx);

  gemm_wmma<0><<<dim3((3 * HID_) / 128, ROWS_ / 128), 128, 0, stream>>>(
      nx, HID_, wqkv_t, HID_, qkvb, nullptr, bufA, nullptr, 3 * HID_, HID_);
  repack_qkv<<<(B_ * NH_ * N_ * DH_) / 256, 256, 0, stream>>>(bufA, qf, kf, vtf);

  bigbird_attn_kernel<<<dim3(M_, NH_, B_), 128, 0, stream>>>(qf, kf, vtf, ra, tom, fromm, ctx);

  gemm_wmma<1><<<dim3(HID_ / 128, ROWS_ / 128), 128, 0, stream>>>(
      ctx, HID_, wo_t, HID_, obias, x, nullptr, attn, HID_, HID_);

  layernorm_f16_kernel<<<ROWS_, 256, 0, stream>>>(attn, ln2g, ln2b, nx);

  gemm_wmma<2><<<dim3(FF_ / 128, ROWS_ / 128), 128, 0, stream>>>(
      nx, HID_, wff1_t, HID_, bff1, nullptr, bufA, nullptr, FF_, HID_);

  gemm_wmma<1><<<dim3(HID_ / 128, ROWS_ / 128), 128, 0, stream>>>(
      bufA, FF_, wff2_t, FF_, bff2, attn, nullptr, out, HID_, FF_);
}
